// LSTMModel2_70677981823288
// MI455X (gfx1250) — matrix operand order repaired and checked by reading, not compiled, not run on hardware
//
#include <hip/hip_runtime.h>

// ---------------- problem constants ----------------
#define NB   256     // batch
#define LL   256     // sequence length
#define HIN  64      // x features (prev feedback handled separately)
#define H1S  512     // hidden size (both LSTM layers)
#define G4   2048    // 4*H1S gate rows
#define H2S  256     // MLP hidden
#define NLH  (NB * LL * H1S)
#define NBLK 32      // persistent blocks (one hidden 16-col tile each)

// ---------------- LDS layout (bytes), padded rows for bank-conflict-free B reads
#define PITCH   520  // halfs per K=512 row (260 dwords; 260%64==4 -> 64-bank spread)
#define PITCHX  72   // halfs per K=64 row  (36 dwords; 36%64 coprime spread)
#define OFF_WIH1X 0                              // 64 rows * 72 h * 2B  =  9216
#define OFF_WHH1  9216                           // 64 rows * 520 * 2   = 66560
#define OFF_WIH2  75776                          // 66560
#define OFF_WHH2  142336                         // 66560
#define OFF_W1    208896                         // 16 rows * 520 * 2   = 16640
#define OFF_ZBUF  225536                         // 256*16*4            = 16384
#define LDS_BYTES 241920                         // < 320KB WGP LDS

// ---------------- WMMA types ----------------
typedef __attribute__((ext_vector_type(16))) __bf16        v16bf;
typedef __attribute__((ext_vector_type(8)))  float         v8f;
typedef __attribute__((ext_vector_type(8)))  unsigned int  u32x8;

__device__ __forceinline__ unsigned short f2bf(float f) {
    unsigned int u = __float_as_uint(f);
    unsigned int r = u + 0x7FFFu + ((u >> 16) & 1u);
    return (unsigned short)(r >> 16);
}
__device__ __forceinline__ float sigmoidf_(float x) { return 1.0f / (1.0f + __expf(-x)); }

// A fragment: 16x32 bf16, lane = row m; k chunks {kb+8*half, kb+16+8*half}
__device__ __forceinline__ v16bf load_frag_a(const unsigned short* row, int kb, int half) {
    const uint4* p0 = reinterpret_cast<const uint4*>(row + kb + 8 * half);
    const uint4* p1 = reinterpret_cast<const uint4*>(row + kb + 16 + 8 * half);
    uint4 a = *p0, b = *p1;
    u32x8 u = {a.x, a.y, a.z, a.w, b.x, b.y, b.z, b.w};
    return __builtin_bit_cast(v16bf, u);
}
// B fragment: 32x16 bf16, lane = col n; k chunks {kb+8*half, kb+16+8*half} (LDS row)
__device__ __forceinline__ v16bf load_frag_b(const unsigned short* row, int kb, int half) {
    const uint4* p0 = reinterpret_cast<const uint4*>(row + kb + 8 * half);
    const uint4* p1 = reinterpret_cast<const uint4*>(row + kb + 16 + 8 * half);
    uint4 a = *p0, b = *p1;
    u32x8 u = {a.x, a.y, a.z, a.w, b.x, b.y, b.z, b.w};
    return __builtin_bit_cast(v16bf, u);
}
__device__ __forceinline__ v8f wmma_bf16(v16bf a, v16bf b, v8f c) {
    return __builtin_amdgcn_wmma_f32_16x16x32_bf16(false, a, false, b, (short)0, c, false, false);
}

// ---------------- device-wide split barrier (arrive counter + generation) ------
__device__ __forceinline__ void grid_barrier(unsigned* bar, unsigned* gen) {
    __threadfence();           // release prior stores to device scope
    __syncthreads();
    if (threadIdx.x == 0) {
        unsigned g = __hip_atomic_load(gen, __ATOMIC_ACQUIRE, __HIP_MEMORY_SCOPE_AGENT);
        unsigned a = __hip_atomic_fetch_add(bar, 1u, __ATOMIC_ACQ_REL, __HIP_MEMORY_SCOPE_AGENT);
        if (a == (unsigned)(NBLK - 1)) {
            __hip_atomic_store(bar, 0u, __ATOMIC_RELAXED, __HIP_MEMORY_SCOPE_AGENT);
            __hip_atomic_store(gen, g + 1u, __ATOMIC_RELEASE, __HIP_MEMORY_SCOPE_AGENT);
        } else {
            while (__hip_atomic_load(gen, __ATOMIC_ACQUIRE, __HIP_MEMORY_SCOPE_AGENT) == g) {
                __builtin_amdgcn_s_sleep(2);
            }
        }
    }
    __syncthreads();
    __threadfence();           // acquire side: invalidate near caches before reads
}

// ---------------- prologue kernels ----------------
__global__ __launch_bounds__(256) void k_f32_to_bf16(const float* __restrict__ src,
                                                     unsigned short* __restrict__ dst, int n) {
    int i = blockIdx.x * 256 + threadIdx.x;
    if (i < n) dst[i] = f2bf(src[i]);
}
// W_ih1 [2048][65] -> bf16 [2048][64] + f32 prev-column [2048]
__global__ __launch_bounds__(256) void k_prep_wih1(const float* __restrict__ W,
                                                   unsigned short* __restrict__ wx,
                                                   float* __restrict__ wp) {
    int i = blockIdx.x * 256 + threadIdx.x;
    if (i < G4 * HIN) {
        int r = i >> 6, c = i & 63;
        wx[i] = f2bf(W[r * 65 + c]);
    }
    if (i < G4) wp[i] = W[i * 65 + 64];
}

// ---------------- persistent LSTM kernel ----------------
__global__ __launch_bounds__(512, 1) void lstm_persistent(
    const unsigned short* __restrict__ xbf,      // [N][L][64]
    const unsigned short* __restrict__ wih1x_g,  // [2048][64]
    const float*          __restrict__ wih1p,    // [2048]
    const unsigned short* __restrict__ whh1_g,   // [2048][512]
    const unsigned short* __restrict__ wih2_g,   // [2048][512]
    const unsigned short* __restrict__ whh2_g,   // [2048][512]
    const unsigned short* __restrict__ w1_g,     // [256][512]
    const float* __restrict__ b_ih1, const float* __restrict__ b_hh1,
    const float* __restrict__ b_ih2, const float* __restrict__ b_hh2,
    const float* __restrict__ b1, const float* __restrict__ w2, const float* __restrict__ b2,
    unsigned short* __restrict__ h1a, unsigned short* __restrict__ h1b,
    float* __restrict__ c1a, float* __restrict__ c1b,
    unsigned short* __restrict__ h2a, unsigned short* __restrict__ h2b,
    float* __restrict__ c2a, float* __restrict__ c2b,
    float* __restrict__ partial,                 // [256][16]
    unsigned* __restrict__ barws,                // [2] = {arrive, generation}
    float* __restrict__ out_y,
    float* __restrict__ hs1, float* __restrict__ cs1,
    float* __restrict__ hs2, float* __restrict__ cs2) {

    extern __shared__ __align__(16) char smem[];
    unsigned short* L_wih1x = (unsigned short*)(smem + OFF_WIH1X);
    unsigned short* L_whh1  = (unsigned short*)(smem + OFF_WHH1);
    unsigned short* L_wih2  = (unsigned short*)(smem + OFF_WIH2);
    unsigned short* L_whh2  = (unsigned short*)(smem + OFF_WHH2);
    unsigned short* L_w1    = (unsigned short*)(smem + OFF_W1);
    float*          L_z     = (float*)(smem + OFF_ZBUF);

    const int tid = threadIdx.x;
    const int blk = blockIdx.x;
    const int j0  = blk << 4;                    // hidden-column tile owned by block

    // ---- stage weight slices into LDS (once) ----
    for (int idx = tid; idx < 64 * 8; idx += 512) {            // wih1x: 64 rows x 8 uint4
        int row = idx >> 3, c = idx & 7;
        int grow = (row >> 4) * H1S + j0 + (row & 15);
        *(uint4*)(L_wih1x + row * PITCHX + c * 8) =
            *(const uint4*)(wih1x_g + (size_t)grow * HIN + c * 8);
    }
    for (int idx = tid; idx < 64 * 64; idx += 512) {           // 64 rows x 64 uint4 each
        int row = idx >> 6, c = idx & 63;
        int grow = (row >> 4) * H1S + j0 + (row & 15);
        size_t gs = (size_t)grow * H1S + c * 8;
        int ls = row * PITCH + c * 8;
        *(uint4*)(L_whh1 + ls) = *(const uint4*)(whh1_g + gs);
        *(uint4*)(L_wih2 + ls) = *(const uint4*)(wih2_g + gs);
        *(uint4*)(L_whh2 + ls) = *(const uint4*)(whh2_g + gs);
    }
    if (blk < 16) {
        for (int idx = tid; idx < 16 * 64; idx += 512) {       // W1 slice: 16 rows
            int row = idx >> 6, c = idx & 63;
            *(uint4*)(L_w1 + row * PITCH + c * 8) =
                *(const uint4*)(w1_g + (size_t)(j0 + row) * H1S + c * 8);
        }
    }
    __syncthreads();

    const int lane = tid & 31;
    const int w    = tid >> 5;                   // wave 0..15
    const int m0   = w << 4;                     // batch tile
    const int nIdx = lane & 15;
    const int half = lane >> 4;
    const int j    = j0 + nIdx;                  // hidden index of this lane

    // loop-invariant epilogue constants
    const float wp0 = wih1p[0 * H1S + j], wp1 = wih1p[1 * H1S + j];
    const float wp2 = wih1p[2 * H1S + j], wp3 = wih1p[3 * H1S + j];
    const float bsA0 = b_ih1[0 * H1S + j] + b_hh1[0 * H1S + j];
    const float bsA1 = b_ih1[1 * H1S + j] + b_hh1[1 * H1S + j];
    const float bsA2 = b_ih1[2 * H1S + j] + b_hh1[2 * H1S + j];
    const float bsA3 = b_ih1[3 * H1S + j] + b_hh1[3 * H1S + j];
    const float bsB0 = b_ih2[0 * H1S + j] + b_hh2[0 * H1S + j];
    const float bsB1 = b_ih2[1 * H1S + j] + b_hh2[1 * H1S + j];
    const float bsB2 = b_ih2[2 * H1S + j] + b_hh2[2 * H1S + j];
    const float bsB3 = b_ih2[3 * H1S + j] + b_hh2[3 * H1S + j];
    const float b2v  = b2[0];
    const float b1v  = (blk < 16) ? b1[j] : 0.0f;

    // loop-invariant LDS B-row pointers (per lane column)
    const unsigned short* Bx[4];  const unsigned short* Bh1[4];
    const unsigned short* Bi2[4]; const unsigned short* Bh2[4];
#pragma unroll
    for (int g = 0; g < 4; ++g) {
        Bx[g]  = L_wih1x + (g * 16 + nIdx) * PITCHX;
        Bh1[g] = L_whh1  + (g * 16 + nIdx) * PITCH;
        Bi2[g] = L_wih2  + (g * 16 + nIdx) * PITCH;
        Bh2[g] = L_whh2  + (g * 16 + nIdx) * PITCH;
    }
    const unsigned short* Bw1 = L_w1 + nIdx * PITCH;

    for (int t = 0; t < LL; ++t) {
        const bool par = (t & 1) != 0;
        const unsigned short* h1p = par ? h1b : h1a;
        unsigned short*       h1n = par ? h1a : h1b;
        const float*          c1p = par ? c1b : c1a;
        float*                c1n = par ? c1a : c1b;
        const unsigned short* h2p = par ? h2b : h2a;
        unsigned short*       h2n = par ? h2a : h2b;
        const float*          c2p = par ? c2b : c2a;
        float*                c2n = par ? c2a : c2b;

        // ================= Stage A: layer-1 =================
        {
            v8f acc0 = {}, acc1 = {}, acc2 = {}, acc3 = {};
            const unsigned short* aRow = xbf + ((size_t)(m0 + nIdx) * LL + t) * HIN;
#pragma unroll
            for (int kb = 0; kb < HIN; kb += 32) {
                v16bf a = load_frag_a(aRow, kb, half);
                acc0 = wmma_bf16(a, load_frag_b(Bx[0], kb, half), acc0);
                acc1 = wmma_bf16(a, load_frag_b(Bx[1], kb, half), acc1);
                acc2 = wmma_bf16(a, load_frag_b(Bx[2], kb, half), acc2);
                acc3 = wmma_bf16(a, load_frag_b(Bx[3], kb, half), acc3);
            }
            const unsigned short* aRowH = h1p + (size_t)(m0 + nIdx) * H1S;
#pragma unroll 4
            for (int kb = 0; kb < H1S; kb += 32) {
                v16bf a = load_frag_a(aRowH, kb, half);
                acc0 = wmma_bf16(a, load_frag_b(Bh1[0], kb, half), acc0);
                acc1 = wmma_bf16(a, load_frag_b(Bh1[1], kb, half), acc1);
                acc2 = wmma_bf16(a, load_frag_b(Bh1[2], kb, half), acc2);
                acc3 = wmma_bf16(a, load_frag_b(Bh1[3], kb, half), acc3);
            }
#pragma unroll
            for (int v = 0; v < 8; ++v) {
                const int m = m0 + v + 8 * half;
                float tot = b2v;                 // reconstruct prev output (deterministic sum)
                const float* pp = partial + m * 16;
#pragma unroll
                for (int b = 0; b < 16; ++b) tot += pp[b];
                const float pv = (t == 0) ? 0.0f : tot;
                float gi = acc0[v] + pv * wp0 + bsA0;
                float gf = acc1[v] + pv * wp1 + bsA1;
                float gg = acc2[v] + pv * wp2 + bsA2;
                float go = acc3[v] + pv * wp3 + bsA3;
                float cold = c1p[(size_t)m * H1S + j];
                float cn = sigmoidf_(gf) * cold + sigmoidf_(gi) * tanhf(gg);
                float hn = sigmoidf_(go) * tanhf(cn);
                c1n[(size_t)m * H1S + j] = cn;
                h1n[(size_t)m * H1S + j] = f2bf(hn);
                size_t o = ((size_t)m * LL + t) * H1S + j;
                hs1[o] = hn;
                cs1[o] = cn;
                if (t > 0 && blk == 0 && nIdx == 0)
                    out_y[(size_t)m * LL + (t - 1)] = tot;
            }
        }
        grid_barrier(barws, barws + 1);

        // ================= Stage B: layer-2 =================
        {
            v8f acc0 = {}, acc1 = {}, acc2 = {}, acc3 = {};
            const unsigned short* aRow1 = h1n + (size_t)(m0 + nIdx) * H1S;   // current h1
#pragma unroll 4
            for (int kb = 0; kb < H1S; kb += 32) {
                v16bf a = load_frag_a(aRow1, kb, half);
                acc0 = wmma_bf16(a, load_frag_b(Bi2[0], kb, half), acc0);
                acc1 = wmma_bf16(a, load_frag_b(Bi2[1], kb, half), acc1);
                acc2 = wmma_bf16(a, load_frag_b(Bi2[2], kb, half), acc2);
                acc3 = wmma_bf16(a, load_frag_b(Bi2[3], kb, half), acc3);
            }
            const unsigned short* aRow2 = h2p + (size_t)(m0 + nIdx) * H1S;
#pragma unroll 4
            for (int kb = 0; kb < H1S; kb += 32) {
                v16bf a = load_frag_a(aRow2, kb, half);
                acc0 = wmma_bf16(a, load_frag_b(Bh2[0], kb, half), acc0);
                acc1 = wmma_bf16(a, load_frag_b(Bh2[1], kb, half), acc1);
                acc2 = wmma_bf16(a, load_frag_b(Bh2[2], kb, half), acc2);
                acc3 = wmma_bf16(a, load_frag_b(Bh2[3], kb, half), acc3);
            }
#pragma unroll
            for (int v = 0; v < 8; ++v) {
                const int m = m0 + v + 8 * half;
                float gi = acc0[v] + bsB0;
                float gf = acc1[v] + bsB1;
                float gg = acc2[v] + bsB2;
                float go = acc3[v] + bsB3;
                float cold = c2p[(size_t)m * H1S + j];
                float cn = sigmoidf_(gf) * cold + sigmoidf_(gi) * tanhf(gg);
                float hn = sigmoidf_(go) * tanhf(cn);
                c2n[(size_t)m * H1S + j] = cn;
                h2n[(size_t)m * H1S + j] = f2bf(hn);
                size_t o = ((size_t)m * LL + t) * H1S + j;
                hs2[o] = hn;
                cs2[o] = cn;
            }
        }
        grid_barrier(barws, barws + 1);

        // ================= Stage C: MLP partials (blocks 0..15) =================
        if (blk < 16) {
            v8f acc = {};
            const unsigned short* aRow = h2n + (size_t)(m0 + nIdx) * H1S;
#pragma unroll 4
            for (int kb = 0; kb < H1S; kb += 32) {
                v16bf a = load_frag_a(aRow, kb, half);
                acc = wmma_bf16(a, load_frag_b(Bw1, kb, half), acc);
            }
#pragma unroll
            for (int v = 0; v < 8; ++v) {
                const int m = m0 + v + 8 * half;
                L_z[m * 16 + nIdx] = fmaxf(acc[v] + b1v, 0.0f);   // relu(h2 W1^T + b1)
            }
            __syncthreads();
            if (tid < NB) {
                const int m = tid;
                const float* zr  = L_z + m * 16;
                const float* w2r = w2 + j0;
                float s = 0.0f;
#pragma unroll
                for (int k2 = 0; k2 < 16; ++k2) s += zr[k2] * w2r[k2];
                partial[m * 16 + blk] = s;       // deterministic: fixed writer per slot
            }
        }
        grid_barrier(barws, barws + 1);
    }

    // tail: out[:, L-1]
    if (blk == 0 && tid < NB) {
        float tot = b2v;
        const float* pp = partial + tid * 16;
#pragma unroll
        for (int b = 0; b < 16; ++b) tot += pp[b];
        out_y[(size_t)tid * LL + (LL - 1)] = tot;
    }
}

// ---------------- host launcher ----------------
extern "C" void kernel_launch(void* const* d_in, const int* in_sizes, int n_in,
                              void* d_out, int out_size, void* d_ws, size_t ws_size,
                              hipStream_t stream) {
    const float* x     = (const float*)d_in[0];
    const float* W_ih1 = (const float*)d_in[1];
    const float* W_hh1 = (const float*)d_in[2];
    const float* b_ih1 = (const float*)d_in[3];
    const float* b_hh1 = (const float*)d_in[4];
    const float* W_ih2 = (const float*)d_in[5];
    const float* W_hh2 = (const float*)d_in[6];
    const float* b_ih2 = (const float*)d_in[7];
    const float* b_hh2 = (const float*)d_in[8];
    const float* W1    = (const float*)d_in[9];
    const float* b1    = (const float*)d_in[10];
    const float* W2    = (const float*)d_in[11];
    const float* b2    = (const float*)d_in[12];

    float* out   = (float*)d_out;
    float* out_y = out;                           // [N][L]
    float* hs1   = out + (size_t)NB * LL;         // [N][L][512]
    float* cs1   = hs1 + (size_t)NLH;
    float* hs2   = cs1 + (size_t)NLH;
    float* cs2   = hs2 + (size_t)NLH;

    char* wsp = (char*)d_ws;
    auto alloc = [&](size_t bytes) -> void* {
        void* p = wsp;
        wsp += (bytes + 255) & ~(size_t)255;
        return p;
    };

    unsigned short* xbf   = (unsigned short*)alloc((size_t)NB * LL * HIN * 2);
    unsigned short* wih1x = (unsigned short*)alloc((size_t)G4 * HIN * 2);
    float*          wih1p = (float*)         alloc((size_t)G4 * 4);
    unsigned short* whh1  = (unsigned short*)alloc((size_t)G4 * H1S * 2);
    unsigned short* wih2  = (unsigned short*)alloc((size_t)G4 * H1S * 2);
    unsigned short* whh2  = (unsigned short*)alloc((size_t)G4 * H1S * 2);
    unsigned short* w1bf  = (unsigned short*)alloc((size_t)H2S * H1S * 2);

    // zero-initialized block: parity-0 states + partials + barrier words (contiguous)
    const size_t zbytes = (size_t)NB * H1S * 2        // h1a
                        + (size_t)NB * H1S * 2        // h2a
                        + (size_t)NB * H1S * 4        // c1a
                        + (size_t)NB * H1S * 4        // c2a
                        + (size_t)NB * 16 * 4         // partial
                        + 256;                        // barrier {arrive, gen}
    char* zblk = (char*)alloc(zbytes);
    unsigned short* h1a;  unsigned short* h2a;  float* c1a;  float* c2a;
    float* partial;  unsigned* barws;
    {
        char* p = zblk;
        h1a     = (unsigned short*)p; p += (size_t)NB * H1S * 2;
        h2a     = (unsigned short*)p; p += (size_t)NB * H1S * 2;
        c1a     = (float*)p;          p += (size_t)NB * H1S * 4;
        c2a     = (float*)p;          p += (size_t)NB * H1S * 4;
        partial = (float*)p;          p += (size_t)NB * 16 * 4;
        barws   = (unsigned*)p;
    }
    unsigned short* h1b = (unsigned short*)alloc((size_t)NB * H1S * 2);
    unsigned short* h2b = (unsigned short*)alloc((size_t)NB * H1S * 2);
    float*          c1b = (float*)alloc((size_t)NB * H1S * 4);
    float*          c2b = (float*)alloc((size_t)NB * H1S * 4);
    (void)ws_size; (void)in_sizes; (void)n_in; (void)out_size;

    // ---- prologue: bf16 conversion + zero init ----
    {
        int n = NB * LL * HIN;
        k_f32_to_bf16<<<(n + 255) / 256, 256, 0, stream>>>(x, xbf, n);
    }
    k_prep_wih1<<<(G4 * HIN + 255) / 256, 256, 0, stream>>>(W_ih1, wih1x, wih1p);
    {
        int n = G4 * H1S;
        k_f32_to_bf16<<<(n + 255) / 256, 256, 0, stream>>>(W_hh1, whh1, n);
        k_f32_to_bf16<<<(n + 255) / 256, 256, 0, stream>>>(W_ih2, wih2, n);
        k_f32_to_bf16<<<(n + 255) / 256, 256, 0, stream>>>(W_hh2, whh2, n);
    }
    {
        int n = H2S * H1S;
        k_f32_to_bf16<<<(n + 255) / 256, 256, 0, stream>>>(W1, w1bf, n);
    }
    hipMemsetAsync(zblk, 0, zbytes, stream);

    // ---- one persistent kernel runs the whole scan ----
    lstm_persistent<<<NBLK, 512, LDS_BYTES, stream>>>(
        xbf, wih1x, wih1p, whh1, wih2, whh2, w1bf,
        b_ih1, b_hh1, b_ih2, b_hh2, b1, W2, b2,
        h1a, h1b, c1a, c1b, h2a, h2b, c2a, c2b,
        partial, barws,
        out_y, hs1, cs1, hs2, cs2);
}
